// ApproachNet_view_fps_objectness_23682449670879
// MI455X (gfx1250) — matrix operand order repaired and checked by reading, not compiled, not run on hardware
//
#include <hip/hip_runtime.h>

#define B_   4
#define N_   20000
#define C_   256
#define V_   300
#define NS_  1024
#define EPSV 1e-5f

typedef __bf16 bf16;
typedef __attribute__((ext_vector_type(16))) __bf16 bf16x16;
typedef __attribute__((ext_vector_type(8)))  float  f32x8;

union bf2u  { bf16 h[2]; unsigned u; };
union frag_u { bf16x16 v; uint4 q[2]; };

__device__ __forceinline__ f32x8 wmma_bf16(bf16x16 a, bf16x16 b, f32x8 c) {
  // D = A(16x32 bf16) * B(32x16 bf16) + C(16x16 f32) -> v_wmma_f32_16x16x32_bf16
  return __builtin_amdgcn_wmma_f32_16x16x32_bf16(false, a, false, b, (short)0, c, false, false);
}

// A-matrix fragment from LDS tile (M-major rows, `stride`=40 bf16 -> 80B, 16B aligned).
// Lane l: M = mbase + (l&15); elements 0..7 -> K=kb..kb+7, 8..15 -> K=kb+16..kb+23,
// kb=(l>>4)*8.  Two contiguous 16B chunks per lane -> 2x ds_load_b128.
__device__ __forceinline__ bf16x16 load_a(const bf16* base, int stride, int mbase, int lane) {
  int m  = mbase + (lane & 15);
  int kb = (lane >> 4) * 8;
  const bf16* p = base + m * stride + kb;
  frag_u u;
  u.q[0] = *(const uint4*)p;
  u.q[1] = *(const uint4*)(p + 16);
  return u.v;
}

// B-matrix fragment from an N-major LDS tile (Xl[n][k], `stride`=40 bf16).
// Lane l: N = nbase + (l&15); elements 0..7 -> K=kb..kb+7, 8..15 -> K=kb+16..kb+23,
// kb=(l>>4)*8.  Two contiguous 16B chunks per lane -> 2x ds_load_b128.
__device__ __forceinline__ bf16x16 load_b_t(const bf16* base, int stride, int nbase, int lane) {
  int n  = nbase + (lane & 15);
  int kb = (lane >> 4) * 8;
  const bf16* p = base + n * stride + kb;
  frag_u u;
  u.q[0] = *(const uint4*)p;
  u.q[1] = *(const uint4*)(p + 16);
  return u.v;
}

// -------- Kernel 1: fused graspable head ------------------------------------
// h = relu(bn(gh_w1 @ feat)) computed in-register (never stored);
// graspable = gh_w2 @ h + gh_b2 reduced across waves via LDS ds_add_f32.
__global__ __launch_bounds__(256)
void k_grasp_head(const float* __restrict__ feat, const float* __restrict__ w1,
                  const float* __restrict__ b1c, const float* __restrict__ gg,
                  const float* __restrict__ gbe, const float* __restrict__ gm,
                  const float* __restrict__ gv, const float* __restrict__ w2h,
                  const float* __restrict__ b2h,
                  float* __restrict__ o_graspness, float* __restrict__ o_objmask,
                  int* __restrict__ ws_mask)
{
  __shared__ __align__(16) bf16  Wl[256 * 40];   // 256 M rows x 32 K (stride 40)
  __shared__ __align__(16) bf16  Xl[64 * 40];    // 64 N rows x 32 K (stride 40)
  __shared__ float sA[256], bA[256], w2s[3 * 256], part[3 * 64];

  int t = threadIdx.x, lane = t & 31, w = t >> 5;
  int b = blockIdx.y;
  int n0 = blockIdx.x * 64;

  {   // per-channel BN folding + head weights into LDS
    float s = gg[t] * rsqrtf(gv[t] + EPSV);
    sA[t] = s;
    bA[t] = (b1c[t] - gm[t]) * s + gbe[t];
    w2s[t] = w2h[t]; w2s[256 + t] = w2h[256 + t]; w2s[512 + t] = w2h[512 + t];
  }
  if (t < 192) part[t] = 0.f;

  f32x8 zero = {0.f, 0.f, 0.f, 0.f, 0.f, 0.f, 0.f, 0.f};
  f32x8 acc[2][4];
#pragma unroll
  for (int i = 0; i < 2; ++i)
#pragma unroll
    for (int j = 0; j < 4; ++j) acc[i][j] = zero;

  const float* fb = feat + (size_t)b * C_ * N_;

  for (int kk = 0; kk < C_; kk += 32) {
    __syncthreads();
    // stage W slab 256x32, packed as b32 pairs (coalesced along K within a row)
#pragma unroll
    for (int i = 0; i < 16; ++i) {
      int e = i * 256 + t;            // 4096 K-pairs
      int r = e >> 4, c = (e & 15) * 2;
      bf2u u;
      u.h[0] = (bf16)w1[r * C_ + kk + c];
      u.h[1] = (bf16)w1[r * C_ + kk + c + 1];
      *(unsigned*)(&Wl[r * 40 + c]) = u.u;
    }
    // stage X slab transposed to N-major: Xl[n][k], packed b32 pairs along K.
    // Global reads stay coalesced along N (two full rows per K-pair).
#pragma unroll
    for (int i = 0; i < 4; ++i) {
      int e = i * 256 + t;            // 1024 (n, K-pair) entries
      int kp = e >> 6, n = e & 63;
      int k = kp * 2;
      int gn = n0 + n;
      float v0 = (gn < N_) ? fb[(size_t)(kk + k) * N_ + gn] : 0.f;
      float v1 = (gn < N_) ? fb[(size_t)(kk + k + 1) * N_ + gn] : 0.f;
      bf2u u; u.h[0] = (bf16)v0; u.h[1] = (bf16)v1;
      *(unsigned*)(&Xl[n * 40 + k]) = u.u;
    }
    __syncthreads();
    bf16x16 a0 = load_a(Wl, 40, w * 32, lane);
    bf16x16 a1 = load_a(Wl, 40, w * 32 + 16, lane);
#pragma unroll
    for (int nt = 0; nt < 4; ++nt) {
      bf16x16 bq = load_b_t(Xl, 40, nt * 16, lane);
      acc[0][nt] = wmma_bf16(a0, bq, acc[0][nt]);
      acc[1][nt] = wmma_bf16(a1, bq, acc[1][nt]);
    }
  }
  __syncthreads();

  int lo = lane & 15, hi = lane >> 4;
#pragma unroll
  for (int mt = 0; mt < 2; ++mt)
#pragma unroll
    for (int nt = 0; nt < 4; ++nt) {
      float p0 = 0.f, p1 = 0.f, p2 = 0.f;
      int n = nt * 16 + lo;
#pragma unroll
      for (int r = 0; r < 8; ++r) {
        int m = w * 32 + mt * 16 + 8 * hi + r;
        float h = fmaxf(acc[mt][nt][r] * sA[m] + bA[m], 0.f);
        p0 += w2s[m] * h; p1 += w2s[256 + m] * h; p2 += w2s[512 + m] * h;
      }
      atomicAdd(&part[n], p0);
      atomicAdd(&part[64 + n], p1);
      atomicAdd(&part[128 + n], p2);
    }
  __syncthreads();

  if (t < 64) {
    int n = n0 + t;
    if (n < N_) {
      float g0 = part[t] + b2h[0];
      float g1v = part[64 + t] + b2h[1];
      float g2v = part[128 + t] + b2h[2];
      int obj = (g1v > g0) ? 1 : 0;
      o_graspness[(size_t)b * N_ + n] = g2v;
      o_objmask[(size_t)b * N_ + n]   = (float)obj;
      ws_mask[(size_t)b * N_ + n]     = (obj && g2v > 0.1f) ? 1 : 0;
    }
  }
}

// -------- Kernel 2: masked farthest-point sampling (sequential scan) --------
__global__ __launch_bounds__(1024)
void k_fps(const float* __restrict__ xyz, const int* __restrict__ mask,
           int* __restrict__ ws_inds, float* __restrict__ o_inds)
{
  __shared__ float rv[1024];
  __shared__ int   ri[1024];
  __shared__ int   s_cur;
  int t = threadIdx.x;
  int b = blockIdx.x;
  const float* xb = xyz + (size_t)b * N_ * 3;
  const int*   mb = mask + (size_t)b * N_;

  float dloc[20];
  int first_m = 0x7fffffff;
#pragma unroll
  for (int k = 0; k < 20; ++k) {
    int n = t + k * 1024;
    int m = (n < N_) ? mb[n] : 0;
    dloc[k] = (n < N_ && m) ? 1e10f : -1.f;
    if (m && n < first_m) first_m = n;
  }
  ri[t] = first_m;
  __syncthreads();
  for (int s = 512; s > 0; s >>= 1) {
    if (t < s) { if (ri[t + s] < ri[t]) ri[t] = ri[t + s]; }
    __syncthreads();
  }
  if (t == 0) s_cur = (ri[0] == 0x7fffffff) ? 0 : ri[0];
  __syncthreads();

  for (int step = 0; step < NS_; ++step) {
    int cur = s_cur;
    if (t == 0) {
      ws_inds[b * NS_ + step] = cur;
      o_inds[b * NS_ + step]  = (float)cur;
    }
    float cx = xb[cur * 3], cy = xb[cur * 3 + 1], cz = xb[cur * 3 + 2];
    float bv = -2.f; int bi = 0;
#pragma unroll
    for (int k = 0; k < 20; ++k) {
      int n = t + k * 1024;
      if (n >= N_) continue;
      float v = dloc[k];
      if (v >= 0.f) {
        float dx = xb[n * 3] - cx, dy = xb[n * 3 + 1] - cy, dz = xb[n * 3 + 2] - cz;
        v = fminf(v, dx * dx + dy * dy + dz * dz);
        dloc[k] = v;
      }
      if (v > bv || (v == bv && n < bi)) { bv = v; bi = n; }
    }
    rv[t] = bv; ri[t] = bi;
    __syncthreads();
    for (int s = 512; s > 0; s >>= 1) {
      if (t < s) {
        float ov = rv[t + s]; int oi = ri[t + s];
        if (ov > rv[t] || (ov == rv[t] && oi < ri[t])) { rv[t] = ov; ri[t] = oi; }
      }
      __syncthreads();
    }
    if (t == 0) s_cur = ri[0];
    __syncthreads();
  }
}

// -------- Kernel 3: gather xyz / features / graspness by FPS indices --------
__global__ __launch_bounds__(256)
void k_gather(const float* __restrict__ xyz, const float* __restrict__ feat,
              const int* __restrict__ ws_inds, const float* __restrict__ o_graspness,
              float* __restrict__ o_gxyz, float* __restrict__ o_gfeat,
              float* __restrict__ o_fp2, bf16* __restrict__ Xg)
{
  int idx = blockIdx.x * 256 + threadIdx.x;
  if (idx >= B_ * 257 * NS_) return;
  int j = idx & (NS_ - 1);
  int c = (idx >> 10) % 257;
  int b = idx / (257 * NS_);
  int ind = ws_inds[b * NS_ + j];
  if (c < 256) {
    float v = feat[((size_t)b * C_ + c) * N_ + ind];
    o_gfeat[((size_t)b * C_ + c) * NS_ + j] = v;
    Xg[((size_t)b * C_ + c) * NS_ + j] = (bf16)v;
  } else {
    const float* p = xyz + ((size_t)b * N_ + ind) * 3;
    float* q = o_gxyz + ((size_t)b * NS_ + j) * 3;
    q[0] = p[0]; q[1] = p[1]; q[2] = p[2];
    o_fp2[b * NS_ + j] = o_graspness[(size_t)b * N_ + ind];
  }
}

// -------- Kernel 4: generic bf16-WMMA GEMM  Y = act(bn(W@X + b)) ------------
// W: MxK f32 row-major; X: Kx1024 bf16 per batch; Yb: Mx1024 bf16; Yt: f32
// transposed (n*V + m) for view_score. M,K zero-padded in staging so EXEC
// stays all-ones at every WMMA.
__global__ __launch_bounds__(256)
void k_gemm(const float* __restrict__ W, const float* __restrict__ bias,
            const float* __restrict__ gg, const float* __restrict__ gbe,
            const float* __restrict__ gm, const float* __restrict__ gv,
            const bf16* __restrict__ X, bf16* __restrict__ Yb,
            float* __restrict__ Yt, int M, int K, int relu)
{
  __shared__ __align__(16) bf16  Wl[128 * 40];   // 128 M rows x 32 K
  __shared__ __align__(16) bf16  Xl[64 * 40];    // 64 N rows x 32 K (N-major)
  __shared__ float sA[128], bA[128];

  int t = threadIdx.x, lane = t & 31, w = t >> 5;
  int mb = blockIdx.x, nb = blockIdx.y, b = blockIdx.z;
  int n0 = nb * 64;

  if (t < 128) {
    int gmr = mb * 128 + t;
    float s = 1.f, ba = 0.f;
    if (gmr < M) {
      ba = bias[gmr];
      if (gg) {
        s = gg[gmr] * rsqrtf(gv[gmr] + EPSV);
        ba = (ba - gm[gmr]) * s + gbe[gmr];
      }
    }
    sA[t] = s; bA[t] = ba;
  }

  f32x8 zero = {0.f, 0.f, 0.f, 0.f, 0.f, 0.f, 0.f, 0.f};
  f32x8 acc[4];
#pragma unroll
  for (int j = 0; j < 4; ++j) acc[j] = zero;

  const bf16* Xb = X + (size_t)b * K * NS_;
  int Kp = (K + 31) & ~31;

  for (int kk = 0; kk < Kp; kk += 32) {
    __syncthreads();
    // stage W slab 128x32, packed b32 pairs
#pragma unroll
    for (int i = 0; i < 8; ++i) {
      int e = i * 256 + t;            // 2048 K-pairs
      int r = e >> 4, c = (e & 15) * 2;
      int gmr = mb * 128 + r;
      int k0 = kk + c, k1 = k0 + 1;
      bf2u u;
      u.h[0] = (bf16)((gmr < M && k0 < K) ? W[(size_t)gmr * K + k0] : 0.f);
      u.h[1] = (bf16)((gmr < M && k1 < K) ? W[(size_t)gmr * K + k1] : 0.f);
      *(unsigned*)(&Wl[r * 40 + c]) = u.u;
    }
    // stage X slab transposed to N-major, packed b32 pairs along K
#pragma unroll
    for (int i = 0; i < 4; ++i) {
      int e = i * 256 + t;
      int kp = e >> 6, n = e & 63;
      int k0 = kk + kp * 2, k1 = k0 + 1;
      bf2u u;
      u.h[0] = (k0 < K) ? Xb[(size_t)k0 * NS_ + n0 + n] : (bf16)0.f;
      u.h[1] = (k1 < K) ? Xb[(size_t)k1 * NS_ + n0 + n] : (bf16)0.f;
      *(unsigned*)(&Xl[n * 40 + kp * 2]) = u.u;
    }
    __syncthreads();
    bf16x16 a = load_a(Wl, 40, w * 16, lane);
#pragma unroll
    for (int nt = 0; nt < 4; ++nt) {
      bf16x16 bq = load_b_t(Xl, 40, nt * 16, lane);
      acc[nt] = wmma_bf16(a, bq, acc[nt]);
    }
  }

  int lo = lane & 15, hi = lane >> 4;
#pragma unroll
  for (int nt = 0; nt < 4; ++nt) {
    int n = n0 + nt * 16 + lo;
#pragma unroll
    for (int r = 0; r < 8; ++r) {
      int ml = w * 16 + 8 * hi + r;
      int gmr = mb * 128 + ml;
      if (gmr < M) {
        float y = acc[nt][r] * sA[ml] + bA[ml];
        if (relu) y = fmaxf(y, 0.f);
        if (Yb) Yb[((size_t)b * M + gmr) * NS_ + n] = (bf16)y;
        if (Yt) Yt[((size_t)b * NS_ + n) * V_ + gmr] = y;
      }
    }
  }
}

// -------- Kernel 5: per-point view argmax + Fibonacci template + rotation ---
__global__ __launch_bounds__(256)
void k_view(const float* __restrict__ vs, float* __restrict__ o_tvi,
            float* __restrict__ o_tvs, float* __restrict__ o_vpxyz,
            float* __restrict__ o_vprot)
{
  int idx = blockIdx.x * 256 + threadIdx.x;
  if (idx >= B_ * NS_) return;
  const float* row = vs + (size_t)idx * V_;
  float bv = row[0]; int bi = 0;
  for (int i = 1; i < V_; ++i) { float v = row[i]; if (v > bv) { bv = v; bi = i; } }
  o_tvs[idx] = bv;
  o_tvi[idx] = (float)bi;

  const float PHI = 0.6180339887498949f;
  float zi = (2.f * bi + 1.f) / (float)V_ - 1.f;
  float rr = sqrtf(fmaxf(1.f - zi * zi, 0.f));
  float ang = 2.f * 3.14159265358979f * (float)bi * PHI;
  float xi = rr * cosf(ang), yi = rr * sinf(ang);
  float* vx = o_vpxyz + (size_t)idx * 3;
  vx[0] = xi; vx[1] = yi; vx[2] = zi;

  float ax = -xi, ay = -yi, az = -zi;          // towards = -vp
  float byx = -ay, byy = ax, byz = 0.f;
  float ny = sqrtf(byx * byx + byy * byy);
  if (ny == 0.f) { byx = 0.f; byy = 1.f; byz = 0.f; ny = 1.f; }
  float nx = sqrtf(ax * ax + ay * ay + az * az);
  ax /= nx; ay /= nx; az /= nx;
  byx /= ny; byy /= ny; byz /= ny;
  float czx = ay * byz - az * byy;
  float czy = az * byx - ax * byz;
  float czz = ax * byy - ay * byx;
  float* R = o_vprot + (size_t)idx * 9;        // columns = axis_x, axis_y, axis_z
  R[0] = ax; R[1] = byx; R[2] = czx;
  R[3] = ay; R[4] = byy; R[5] = czy;
  R[6] = az; R[7] = byz; R[8] = czz;
}

extern "C" void kernel_launch(void* const* d_in, const int* in_sizes, int n_in,
                              void* d_out, int out_size, void* d_ws, size_t ws_size,
                              hipStream_t stream)
{
  (void)in_sizes; (void)n_in; (void)out_size; (void)ws_size;
  const float* seed_xyz      = (const float*)d_in[0];
  const float* seed_features = (const float*)d_in[1];
  const float* gh_w1 = (const float*)d_in[2];
  const float* gh_b1 = (const float*)d_in[3];
  const float* gh_g1 = (const float*)d_in[4];
  const float* gh_be1 = (const float*)d_in[5];
  const float* gh_m1 = (const float*)d_in[6];
  const float* gh_v1 = (const float*)d_in[7];
  const float* gh_w2 = (const float*)d_in[8];
  const float* gh_b2 = (const float*)d_in[9];
  const float* w1 = (const float*)d_in[10];
  const float* b1 = (const float*)d_in[11];
  const float* g1 = (const float*)d_in[12];
  const float* be1 = (const float*)d_in[13];
  const float* m1 = (const float*)d_in[14];
  const float* v1 = (const float*)d_in[15];
  const float* w2 = (const float*)d_in[16];
  const float* b2 = (const float*)d_in[17];
  const float* g2 = (const float*)d_in[18];
  const float* be2 = (const float*)d_in[19];
  const float* m2 = (const float*)d_in[20];
  const float* v2 = (const float*)d_in[21];
  const float* w3 = (const float*)d_in[22];
  const float* b3 = (const float*)d_in[23];

  // Output tuple, concatenated flat in return order.
  float* out = (float*)d_out;
  float* o_graspness = out;                               // B*N
  float* o_objmask   = o_graspness + (size_t)B_ * N_;     // B*N
  float* o_gxyz      = o_objmask + (size_t)B_ * N_;       // B*NS*3
  float* o_inds      = o_gxyz + (size_t)B_ * NS_ * 3;     // B*NS
  float* o_gfeat     = o_inds + (size_t)B_ * NS_;         // B*C*NS
  float* o_fp2       = o_gfeat + (size_t)B_ * C_ * NS_;   // B*NS
  float* o_vscore    = o_fp2 + (size_t)B_ * NS_;          // B*NS*V
  float* o_tvi       = o_vscore + (size_t)B_ * NS_ * V_;  // B*NS
  float* o_tvs       = o_tvi + (size_t)B_ * NS_;          // B*NS
  float* o_vpxyz     = o_tvs + (size_t)B_ * NS_;          // B*NS*3
  float* o_vprot     = o_vpxyz + (size_t)B_ * NS_ * 3;    // B*NS*9

  // Workspace layout (256B aligned chunks).
  char* ws = (char*)d_ws;
  auto align256 = [](size_t x) { return (x + 255) & ~(size_t)255; };
  int* ws_mask = (int*)ws;  ws += align256((size_t)B_ * N_ * sizeof(int));
  int* ws_inds = (int*)ws;  ws += align256((size_t)B_ * NS_ * sizeof(int));
  bf16* ws_Xg = (bf16*)ws;  ws += align256((size_t)B_ * C_ * NS_ * sizeof(bf16));
  bf16* ws_f1 = (bf16*)ws;  ws += align256((size_t)B_ * C_ * NS_ * sizeof(bf16));
  bf16* ws_f2 = (bf16*)ws;  ws += align256((size_t)B_ * V_ * NS_ * sizeof(bf16));

  // 1) graspable head (WMMA GEMM fused with BN/ReLU + 3-ch head + masks)
  k_grasp_head<<<dim3((N_ + 63) / 64, B_), 256, 0, stream>>>(
      seed_features, gh_w1, gh_b1, gh_g1, gh_be1, gh_m1, gh_v1, gh_w2, gh_b2,
      o_graspness, o_objmask, ws_mask);

  // 2) masked FPS (sequential, one workgroup per batch)
  k_fps<<<dim3(B_), 1024, 0, stream>>>(seed_xyz, ws_mask, ws_inds, o_inds);

  // 3) gather sampled xyz / features / graspness (+ bf16 copy for GEMMs)
  k_gather<<<dim3((B_ * 257 * NS_ + 255) / 256), 256, 0, stream>>>(
      seed_xyz, seed_features, ws_inds, o_graspness, o_gxyz, o_gfeat, o_fp2, ws_Xg);

  // 4) view-score MLP: 256->256 (bn+relu), 256->300 (bn+relu), 300->300 (+b)
  k_gemm<<<dim3(2, 16, B_), 256, 0, stream>>>(w1, b1, g1, be1, m1, v1,
                                              ws_Xg, ws_f1, nullptr, 256, 256, 1);
  k_gemm<<<dim3(3, 16, B_), 256, 0, stream>>>(w2, b2, g2, be2, m2, v2,
                                              ws_f1, ws_f2, nullptr, 300, 256, 1);
  k_gemm<<<dim3(3, 16, B_), 256, 0, stream>>>(w3, b3, nullptr, nullptr, nullptr, nullptr,
                                              ws_f2, nullptr, o_vscore, 300, 300, 0);

  // 5) per-point view argmax + template + rotation matrices
  k_view<<<dim3((B_ * NS_ + 255) / 256), 256, 0, stream>>>(
      o_vscore, o_tvi, o_tvs, o_vpxyz, o_vprot);
}
